// HyperGraphAttentionLayerSparse_44504451121419
// MI455X (gfx1250) — compile-verified, matrix operand order checked by reading, not run on hardware
//
#include <hip/hip_runtime.h>
#include <math.h>

// ---------------- problem constants ----------------
#define N_EDGE 4000
#define N_NODE 10000
#define D_IN   256
#define D_OUT  128
#define ALPHA     0.2f
#define OUT_SLOPE 0.01f

typedef __attribute__((ext_vector_type(16))) _Float16 v16h;
typedef __attribute__((ext_vector_type(8)))  float    v8f;

// K index held by element i of lane `lane` in a 16-bit A/B WMMA fragment
// (CDNA5 ISA 7.12.2, 16-bit A-matrix 16x32 layout; B treated column-major
// with the same K striping and lane&15 -> N).
__device__ __forceinline__ int wmma_k(int i, int lane) {
  int p = i >> 1, off = i & 1;
  int kbase = (p < 4) ? (p * 2) : (16 + (p - 4) * 2);
  return kbase + off + ((lane >> 4) << 3);
}

__device__ __forceinline__ float leaky(float v, float s) { return v > 0.f ? v : s * v; }

// order-preserving float<->uint encoding so atomicMax(uint) == float max
__device__ __forceinline__ unsigned f2ord(float f) {
  unsigned u = __float_as_uint(f);
  return (u & 0x80000000u) ? ~u : (u | 0x80000000u);
}
__device__ __forceinline__ float ord2f(unsigned u) {
  return __uint_as_float((u & 0x80000000u) ? (u & 0x7fffffffu) : ~u);
}

// ---------------- tiny prep kernels ----------------
// v2[k] = sum_d weight2[k,d]*a2[d]   (k < 256)
// v3[k] = sum_d weight3[k,d]*a2[128+d] (k < 128)
__global__ void k_prep_vecs(const float* __restrict__ w2, const float* __restrict__ w3,
                            const float* __restrict__ a2, float* v2, float* v3) {
  int t = blockIdx.x * blockDim.x + threadIdx.x;
  if (t < D_IN) {
    float s = 0.f;
    for (int d = 0; d < D_OUT; ++d) s += w2[t * D_OUT + d] * a2[d];
    v2[t] = s;
  } else if (t < D_IN + D_OUT) {
    int k = t - D_IN;
    float s = 0.f;
    for (int d = 0; d < D_OUT; ++d) s += w3[k * D_OUT + d] * a2[D_OUT + d];
    v3[k] = s;
  }
}

__global__ void k_init(unsigned* mEnc, float* Z, float* cn) {
  int t = blockIdx.x * blockDim.x + threadIdx.x;
  if (t < N_NODE) { mEnc[t] = 0u; Z[t] = 0.f; }
  if (t < D_OUT)  cn[t] = 0.f;
}

// s_node[n] = x[n,:] . v2
__global__ void k_snode(const float* __restrict__ x, const float* __restrict__ v2,
                        float* s_node) {
  int n = blockIdx.x * blockDim.x + threadIdx.x;
  if (n >= N_NODE) return;
  float s = 0.f;
  for (int k = 0; k < D_IN; ++k) s += x[n * D_IN + k] * v2[k];
  s_node[n] = s;
}

// ---------------- xt = x @ weight  (WMMA) ----------------
// 625 blocks (16-row tiles) x 256 threads (8 waves = 8 column tiles of 16)
__global__ void __launch_bounds__(256) k_xt_gemm(const float* __restrict__ x,
                                                 const float* __restrict__ w,
                                                 float* __restrict__ xt) {
  __shared__ _Float16 As[16][32];
  int wv = threadIdx.x >> 5, ln = threadIdx.x & 31;
  int m0 = blockIdx.x * 16, d0 = wv * 16;
  v8f c = {};
  for (int k0 = 0; k0 < D_IN; k0 += 32) {
    int id = threadIdx.x;
#pragma unroll
    for (int r = 0; r < 2; ++r, id += 256) {
      int e = id >> 5, kk = id & 31;                 // coalesced along K
      As[e][kk] = (_Float16)x[(size_t)(m0 + e) * D_IN + k0 + kk];
    }
    __syncthreads();
    v16h a, b;
#pragma unroll
    for (int i = 0; i < 16; ++i) {
      int kk = wmma_k(i, ln);
      a[i] = As[ln & 15][kk];
      b[i] = (_Float16)w[(k0 + kk) * D_OUT + d0 + (ln & 15)];
    }
    c = __builtin_amdgcn_wmma_f32_16x16x32_f16(false, a, false, b, (short)0, c, false, false);
    __syncthreads();
  }
#pragma unroll
  for (int v = 0; v < 8; ++v) {
    int m = m0 + v + ((ln >> 4) << 3);
    xt[(size_t)m * D_OUT + d0 + (ln & 15)] = c[v];
  }
}

// ---------------- att_edge: softmax(where(adj>0,1,-9e15)) == mask/deg ----------------
__global__ void __launch_bounds__(256) k_att_edge(const float* __restrict__ adj,
                                                  float* __restrict__ attE) {
  __shared__ float red[8];
  __shared__ float s_inv;
  __shared__ int   s_deg;
  int e = blockIdx.x;
  const float* row = adj + (size_t)e * N_NODE;
  float cnt = 0.f;
  for (int n = threadIdx.x; n < N_NODE; n += 256) cnt += (row[n] > 0.f) ? 1.f : 0.f;
  for (int off = 16; off > 0; off >>= 1) cnt += __shfl_down(cnt, off, 32);
  if ((threadIdx.x & 31) == 0) red[threadIdx.x >> 5] = cnt;
  __syncthreads();
  if (threadIdx.x == 0) {
    float t = 0.f;
    for (int i = 0; i < 8; ++i) t += red[i];
    int deg = (int)(t + 0.5f);
    s_deg = deg;
    s_inv = (deg > 0) ? 1.f / (float)deg : 1e-4f;    // deg==0 -> uniform 1/N_NODE
  }
  __syncthreads();
  float inv = s_inv; int deg = s_deg;
  float* orow = attE + (size_t)e * N_NODE;
  for (int n = threadIdx.x; n < N_NODE; n += 256)
    orow[n] = (deg > 0) ? ((row[n] > 0.f) ? inv : 0.f) : 1e-4f;
}

// ---------------- edge = att_edge @ xt  (WMMA, K=10000 with tail guard) ----------------
__global__ void __launch_bounds__(256) k_edge_gemm(const float* __restrict__ attE,
                                                   const float* __restrict__ xt,
                                                   float* __restrict__ edge) {
  __shared__ _Float16 As[16][32];
  int wv = threadIdx.x >> 5, ln = threadIdx.x & 31;
  int e0 = blockIdx.x * 16, d0 = wv * 16;
  v8f c = {};
  for (int k0 = 0; k0 < N_NODE; k0 += 32) {
    int id = threadIdx.x;
#pragma unroll
    for (int r = 0; r < 2; ++r, id += 256) {
      int ee = id >> 5, kk = id & 31;
      int k = k0 + kk;
      As[ee][kk] = (k < N_NODE) ? (_Float16)attE[(size_t)(e0 + ee) * N_NODE + k]
                                : (_Float16)0.f;
    }
    __syncthreads();
    v16h a, b;
#pragma unroll
    for (int i = 0; i < 16; ++i) {
      int kk = wmma_k(i, ln);
      int k = k0 + kk;
      a[i] = As[ln & 15][kk];
      b[i] = (k < N_NODE) ? (_Float16)xt[(size_t)k * D_OUT + d0 + (ln & 15)] : (_Float16)0.f;
    }
    c = __builtin_amdgcn_wmma_f32_16x16x32_f16(false, a, false, b, (short)0, c, false, false);
    __syncthreads();
  }
#pragma unroll
  for (int v = 0; v < 8; ++v) {
    int ee = e0 + v + ((ln >> 4) << 3);
    edge[(size_t)ee * D_OUT + d0 + (ln & 15)] = c[v];
  }
}

// s_edge[e] = edge[e,:] . v3
__global__ void k_sedge(const float* __restrict__ edge, const float* __restrict__ v3,
                        float* s_edge) {
  int e = blockIdx.x * blockDim.x + threadIdx.x;
  if (e >= N_EDGE) return;
  float s = 0.f;
  for (int d = 0; d < D_OUT; ++d) s += edge[(size_t)e * D_OUT + d] * v3[d];
  s_edge[e] = s;
}

// ---------------- node-softmax pass 1: per-node max over member edges ----------------
__global__ void k_smax(const float* __restrict__ adj, const float* __restrict__ s_edge,
                       const float* __restrict__ s_node, unsigned* mEnc) {
  const size_t total = (size_t)N_EDGE * N_NODE;
  for (size_t idx = (size_t)blockIdx.x * blockDim.x + threadIdx.x; idx < total;
       idx += (size_t)gridDim.x * blockDim.x) {
    float a = adj[idx];
    if (a > 0.f) {
      int e = (int)(idx / N_NODE), n = (int)(idx % N_NODE);
      float val = leaky(s_edge[e] + s_node[n], ALPHA);
      atomicMax(&mEnc[n], f2ord(val));
    }
  }
}

// ---------------- node-softmax pass 2: per-node sum of exp(val - max) ----------------
__global__ void k_ssum(const float* __restrict__ adj, const float* __restrict__ s_edge,
                       const float* __restrict__ s_node, const unsigned* __restrict__ mEnc,
                       float* Z) {
  const size_t total = (size_t)N_EDGE * N_NODE;
  for (size_t idx = (size_t)blockIdx.x * blockDim.x + threadIdx.x; idx < total;
       idx += (size_t)gridDim.x * blockDim.x) {
    float a = adj[idx];
    if (a > 0.f) {
      int e = (int)(idx / N_NODE), n = (int)(idx % N_NODE);
      float val = leaky(s_edge[e] + s_node[n], ALPHA);
      atomicAdd(&Z[n], expf(val - ord2f(mEnc[n])));
    }
  }
}

// ---------------- node = softmax(P) @ edge, fused WMMA; leaky applied on store --------
__global__ void __launch_bounds__(256) k_node_gemm(const float* __restrict__ adj,
                                                   const float* __restrict__ edge,
                                                   const float* __restrict__ s_edge,
                                                   const float* __restrict__ s_node,
                                                   const unsigned* __restrict__ mEnc,
                                                   const float* __restrict__ Z,
                                                   float* __restrict__ node_out) {
  __shared__ _Float16 Ps[32][16];      // probability tile P[e_local][n_local]
  __shared__ float sn_s[16], mm_s[16], iz_s[16];
  int wv = threadIdx.x >> 5, ln = threadIdx.x & 31;
  int n0 = blockIdx.x * 16, d0 = wv * 16;
  if (threadIdx.x < 16) {
    int n = n0 + threadIdx.x;
    float z = Z[n];
    sn_s[threadIdx.x] = s_node[n];
    mm_s[threadIdx.x] = ord2f(mEnc[n]);
    iz_s[threadIdx.x] = (z > 0.f) ? 1.f / z : -1.f;  // <0 flags "no member -> uniform"
  }
  __syncthreads();
  v8f c = {};
  for (int e0 = 0; e0 < N_EDGE; e0 += 32) {          // 4000/32 == 125 exact
    int id = threadIdx.x;
#pragma unroll
    for (int r = 0; r < 2; ++r, id += 256) {
      int el = id >> 4, nl = id & 15;                // 16 consecutive n per 16 threads
      float a = adj[(size_t)(e0 + el) * N_NODE + n0 + nl];
      float iz = iz_s[nl];
      float p;
      if (iz < 0.f) p = 1.f / (float)N_EDGE;         // empty node -> uniform softmax
      else p = (a > 0.f)
                 ? expf(leaky(s_edge[e0 + el] + sn_s[nl], ALPHA) - mm_s[nl]) * iz
                 : 0.f;
      Ps[el][nl] = (_Float16)p;
    }
    __syncthreads();
    v16h a, b;
#pragma unroll
    for (int i = 0; i < 16; ++i) {
      int kk = wmma_k(i, ln);
      a[i] = Ps[kk][ln & 15];
      b[i] = (_Float16)edge[(size_t)(e0 + kk) * D_OUT + d0 + (ln & 15)];
    }
    c = __builtin_amdgcn_wmma_f32_16x16x32_f16(false, a, false, b, (short)0, c, false, false);
    __syncthreads();
  }
#pragma unroll
  for (int v = 0; v < 8; ++v) {
    int n = n0 + v + ((ln >> 4) << 3);
    node_out[(size_t)n * D_OUT + d0 + (ln & 15)] = leaky(c[v], OUT_SLOPE);
  }
}

// ---------------- column-wise L2 norm over node axis (F.normalize dim=1) -------------
__global__ void __launch_bounds__(256) k_colnorm(const float* __restrict__ node, float* cn) {
  __shared__ float ls[D_OUT];
  for (int d = threadIdx.x; d < D_OUT; d += 256) ls[d] = 0.f;
  __syncthreads();
  const size_t total = (size_t)N_NODE * D_OUT;
  for (size_t idx = (size_t)blockIdx.x * blockDim.x + threadIdx.x; idx < total;
       idx += (size_t)gridDim.x * blockDim.x) {
    float y = node[idx];
    atomicAdd(&ls[(int)(idx & (D_OUT - 1))], y * y);  // ds_add_f32
  }
  __syncthreads();
  for (int d = threadIdx.x; d < D_OUT; d += 256) atomicAdd(&cn[d], ls[d]);
}

__global__ void k_normalize(float* node, const float* __restrict__ cn) {
  int idx = blockIdx.x * blockDim.x + threadIdx.x;
  if (idx >= N_NODE * D_OUT) return;
  float nrm = sqrtf(cn[idx & (D_OUT - 1)]);
  node[idx] = node[idx] / fmaxf(nrm, 1e-12f);
}

// ---------------- launcher ----------------
extern "C" void kernel_launch(void* const* d_in, const int* in_sizes, int n_in,
                              void* d_out, int out_size, void* d_ws, size_t ws_size,
                              hipStream_t stream) {
  const float* x   = (const float*)d_in[0];   // [10000,256]
  const float* adj = (const float*)d_in[1];   // [4000,10000]
  const float* w1  = (const float*)d_in[2];   // weight  [256,128]
  const float* w2  = (const float*)d_in[3];   // weight2 [256,128]
  const float* w3  = (const float*)d_in[4];   // weight3 [128,128]
  const float* a2  = (const float*)d_in[5];   // [256]

  float* node_out = (float*)d_out;                                  // [10000,128]
  float* attE     = node_out + (size_t)N_NODE * D_OUT;              // [4000,10000]

  // workspace layout (floats); ~7.3 MB total
  float* ws = (float*)d_ws;
  size_t o = 0;
  float*    xt     = ws + o; o += (size_t)N_NODE * D_OUT;   // 1,280,000
  float*    edge   = ws + o; o += (size_t)N_EDGE * D_OUT;   //   512,000
  float*    s_node = ws + o; o += N_NODE;
  float*    s_edge = ws + o; o += N_EDGE;
  float*    v2     = ws + o; o += D_IN;
  float*    v3     = ws + o; o += D_OUT;
  unsigned* mEnc   = (unsigned*)(ws + o); o += N_NODE;
  float*    Z      = ws + o; o += N_NODE;
  float*    cn     = ws + o; o += D_OUT;

  k_prep_vecs<<<2, 256, 0, stream>>>(w2, w3, a2, v2, v3);
  k_init<<<(N_NODE + 255) / 256, 256, 0, stream>>>(mEnc, Z, cn);
  k_snode<<<(N_NODE + 255) / 256, 256, 0, stream>>>(x, v2, s_node);
  k_xt_gemm<<<N_NODE / 16, 256, 0, stream>>>(x, w1, xt);
  k_att_edge<<<N_EDGE, 256, 0, stream>>>(adj, attE);
  k_edge_gemm<<<N_EDGE / 16, 256, 0, stream>>>(attE, xt, edge);
  k_sedge<<<(N_EDGE + 255) / 256, 256, 0, stream>>>(edge, v3, s_edge);
  k_smax<<<4096, 256, 0, stream>>>(adj, s_edge, s_node, mEnc);
  k_ssum<<<4096, 256, 0, stream>>>(adj, s_edge, s_node, mEnc, Z);
  k_node_gemm<<<N_NODE / 16, 256, 0, stream>>>(adj, edge, s_edge, s_node, mEnc, Z, node_out);
  k_colnorm<<<256, 256, 0, stream>>>(node_out, cn);
  k_normalize<<<(N_NODE * D_OUT + 255) / 256, 256, 0, stream>>>(node_out, cn);

  (void)in_sizes; (void)n_in; (void)out_size; (void)ws_size;
}
